// SegmentedAttention_46102178955634
// MI455X (gfx1250) — matrix operand order repaired and checked by reading, not compiled, not run on hardware
//
#include <hip/hip_runtime.h>

// ---------------------------------------------------------------------------
// Segmented attention for MI455X (gfx1250, wave32, WMMA bf16 path)
//  - bf16 WMMA GEMMs with double-buffered LDS fed by async global->LDS copies
// ---------------------------------------------------------------------------

typedef __attribute__((ext_vector_type(16))) __bf16 v16bf;
typedef __attribute__((ext_vector_type(8)))  __bf16 v8bf;
typedef __attribute__((ext_vector_type(8)))  float  v8f;
typedef int v4i_vec __attribute__((__vector_size__(16)));

#define DIMX   2048
#define HEADS  32
#define DH     64
#define PTOK   16
#define NQKV   6144     // 3*HEADS*DH
#define WSEG   64       // tokens per segment (8192/128)
#define NTOK   16384    // 2 * 8192 total tokens
#define KEYS   80       // PTOK + WSEG
#define KEYPAD 96       // padded to multiple of 32 for bf16 WMMA K

#if __has_builtin(__builtin_amdgcn_global_load_async_to_lds_b128)
#define HAVE_ASYNC_LDS 1
#endif

__device__ __forceinline__ __bf16 f2bf(float f) {
  unsigned u = __builtin_bit_cast(unsigned, f);
  unsigned r = (u + 0x7FFFu + ((u >> 16) & 1u)) >> 16;
  unsigned short s = (unsigned short)r;
  return __builtin_bit_cast(__bf16, s);
}

// 16-byte chunk copy global -> LDS (async on CDNA5 when builtin available)
__device__ __forceinline__ void cp16(const __bf16* g, __bf16* l) {
#if defined(HAVE_ASYNC_LDS)
  __builtin_amdgcn_global_load_async_to_lds_b128(
      (__attribute__((address_space(1))) v4i_vec*)(uintptr_t)g,
      (__attribute__((address_space(3))) v4i_vec*)(unsigned)(uintptr_t)l, 0, 0);
#else
  *(uint4*)l = *(const uint4*)g;
#endif
}

__device__ __forceinline__ void wait_async_all() {
#if defined(HAVE_ASYNC_LDS)
#if __has_builtin(__builtin_amdgcn_s_wait_asynccnt)
  __builtin_amdgcn_s_wait_asynccnt(0);
#else
  asm volatile("s_wait_asynccnt 0" ::: "memory");
#endif
#endif
}

// --------------------------- RMSNorm + cast --------------------------------
__global__ __launch_bounds__(256) void rmsnorm_cast_kernel(
    const float* __restrict__ x, const float* __restrict__ g,
    __bf16* __restrict__ out) {
  const int tok = blockIdx.x;
  const float* row = x + (size_t)tok * DIMX;
  float ss = 0.f;
#pragma unroll
  for (int i = 0; i < DIMX / 256; ++i) {
    float v = row[threadIdx.x + i * 256];
    ss += v * v;
  }
  __shared__ float sred[256];
  sred[threadIdx.x] = ss;
  __syncthreads();
  for (int s = 128; s > 0; s >>= 1) {
    if (threadIdx.x < s) sred[threadIdx.x] += sred[threadIdx.x + s];
    __syncthreads();
  }
  const float inv = rsqrtf(sred[0] / (float)DIMX + 1.1920929e-07f);
  __bf16* orow = out + (size_t)tok * DIMX;
#pragma unroll
  for (int i = 0; i < DIMX / 256; ++i) {
    int c = threadIdx.x + i * 256;
    orow[c] = f2bf(row[c] * inv * g[c]);
  }
}

// --------------------------- f32 -> bf16 cast ------------------------------
__global__ __launch_bounds__(256) void cast_kernel(
    const float* __restrict__ in, __bf16* __restrict__ out, int n) {
  int i = blockIdx.x * 256 + threadIdx.x;
  if (i < n) out[i] = f2bf(in[i]);
}

// --------------------------- WMMA tiled GEMM -------------------------------
// C(MxN) = A(MxK) @ B(KxN), bf16 in, f32 accumulate. Block tile 128x128x64,
// double-buffered LDS with async global->LDS copies. 8 waves (256 thr), each
// wave owns a 32x64 sub-tile = 2x4 WMMA accumulators.
#define BK 64

template <bool BF16OUT>
__global__ __launch_bounds__(256) void gemm_bf16(
    const __bf16* __restrict__ A, const __bf16* __restrict__ B,
    void* __restrict__ Cp, int M, int N, int K) {
  __shared__ __align__(32) __bf16 sA[2][128 * BK];  // 2 x 16 KB
  __shared__ __align__(32) __bf16 sB[2][BK * 128];  // 2 x 16 KB

  const int tid  = threadIdx.x;
  const int lane = tid & 31;
  const int wave = tid >> 5;
  const int half = lane >> 4;
  const int l    = lane & 15;
  const int wm   = wave >> 1;  // 0..3 (M)
  const int wn   = wave & 1;   // 0..1 (N)
  const int mblk = blockIdx.y * 128;
  const int nblk = blockIdx.x * 128;

  const v8f vzero = {};
  v8f acc[2][4];
#pragma unroll
  for (int i = 0; i < 2; ++i)
#pragma unroll
    for (int j = 0; j < 4; ++j) acc[i][j] = vzero;

  // tile loader: A 128xBK (8 uint4/row), B BKx128 (16 uint4/row)
  auto load_tile = [&](int k0, int buf) {
#pragma unroll
    for (int i = 0; i < 4; ++i) {
      int chunk = tid + i * 256;           // 0..1023
      int row = chunk >> 3, c = chunk & 7; // 128 rows x 8 chunks
      cp16(A + (size_t)(mblk + row) * K + k0 + c * 8,
           &sA[buf][row * BK + c * 8]);
    }
#pragma unroll
    for (int i = 0; i < 4; ++i) {
      int chunk = tid + i * 256;            // 0..1023
      int row = chunk >> 4, c = chunk & 15; // BK rows x 16 chunks
      cp16(B + (size_t)(k0 + row) * N + nblk + c * 8,
           &sB[buf][row * 128 + c * 8]);
    }
  };

  const int nk = K / BK;
  load_tile(0, 0);
  int cur = 0;
  for (int kt = 0; kt < nk; ++kt) {
    wait_async_all();
    __syncthreads();  // tile kt resident; previous compute reads retired
    if (kt + 1 < nk) load_tile((kt + 1) * BK, cur ^ 1);
    if (kt + 2 < nk) {
      __builtin_prefetch(A + (size_t)(mblk + (tid >> 1)) * K + (kt + 2) * BK, 0, 1);
      __builtin_prefetch(B + (size_t)((kt + 2) * BK + (tid & 63)) * N + nblk, 0, 1);
    }

    // compute on buffer `cur`: two 32-deep sub-steps
#pragma unroll
    for (int ks = 0; ks < 2; ++ks) {
      // A fragments: lane l holds row M=l; lanes 0-15 take K{0..7,16..23},
      // lanes 16-31 take K{8..15,24..31}  (ISA 16-bit A 16x32 layout)
      v16bf af[2];
#pragma unroll
      for (int i = 0; i < 2; ++i) {
        const __bf16* rp = &sA[cur][(wm * 32 + i * 16 + l) * BK + ks * 32];
        v8bf lo = *(const v8bf*)(rp + half * 8);
        v8bf hi = *(const v8bf*)(rp + 16 + half * 8);
        af[i] = __builtin_shufflevector(lo, hi, 0, 1, 2, 3, 4, 5, 6, 7, 8, 9,
                                        10, 11, 12, 13, 14, 15);
      }
      // B fragments: lane holds column N=l; K in same table order as A
#pragma unroll
      for (int j = 0; j < 4; ++j) {
        v16bf bfrag;
#pragma unroll
        for (int t = 0; t < 16; ++t)
          bfrag[t] = sB[cur][(ks * 32 + (t & 7) + 8 * half + 16 * (t >> 3)) * 128 + wn * 64 + j * 16 + l];
#pragma unroll
        for (int i = 0; i < 2; ++i)
          acc[i][j] = __builtin_amdgcn_wmma_f32_16x16x32_bf16(
              false, af[i], false, bfrag, (short)0, acc[i][j], false, false);
      }
    }
    cur ^= 1;
  }

  // C layout: VGPR v -> M = v + 8*half, N = lane&15
#pragma unroll
  for (int i = 0; i < 2; ++i) {
#pragma unroll
    for (int j = 0; j < 4; ++j) {
      int r0 = mblk + wm * 32 + i * 16 + 8 * half;
      int c0 = nblk + wn * 64 + j * 16 + l;
#pragma unroll
      for (int v = 0; v < 8; ++v) {
        if constexpr (BF16OUT)
          ((__bf16*)Cp)[(size_t)(r0 + v) * N + c0] = f2bf(acc[i][j][v]);
        else
          ((float*)Cp)[(size_t)(r0 + v) * N + c0] = acc[i][j][v];
      }
    }
  }
}

// --------------------------- Attention kernel ------------------------------
// One workgroup per (head, segment). 4 waves, each owns 16 query rows.
__global__ __launch_bounds__(128) void attn_kernel(
    const __bf16* __restrict__ qkv, const __bf16* __restrict__ pm_bf,
    __bf16* __restrict__ aout) {
  const int h = blockIdx.x;
  const int s = blockIdx.y;
  const int tid  = threadIdx.x;
  const int lane = tid & 31;
  const int wave = tid >> 5;
  const int half = lane >> 4;
  const int l    = lane & 15;

  __shared__ __align__(32) __bf16 sKT[DH * KEYS];      // K^T: [dim][key]
  __shared__ __align__(32) __bf16 sV[KEYPAD * DH];     // V:   [key][dim], pad 0
  __shared__ __align__(32) __bf16 sP[4][16 * KEYPAD];  // probs per wave

  const __bf16* pmk = pm_bf + (size_t)h * PTOK * DH;
  const __bf16* pmv = pm_bf + (size_t)(HEADS + h) * PTOK * DH;

  // V (96x64) as uint4: rows 0..15 = pmv, 16..79 = seq v, 80..95 = zero
  for (int c = tid; c < (KEYPAD * DH) / 8; c += 128) {
    int key = c >> 3, q4 = c & 7;
    uint4 val;
    if (key < PTOK)
      val = *(const uint4*)(pmv + key * DH + q4 * 8);
    else if (key < KEYS)
      val = *(const uint4*)(qkv + (size_t)(s * WSEG + key - PTOK) * NQKV +
                            2 * HEADS * DH + h * DH + q4 * 8);
    else
      val = make_uint4(0, 0, 0, 0);
    *(uint4*)(sV + key * DH + q4 * 8) = val;
  }
  // K^T (scalar transpose into LDS)
  for (int e = tid; e < KEYS * DH; e += 128) {
    int key = e >> 6, dim = e & 63;
    __bf16 v;
    if (key < PTOK)
      v = pmk[key * DH + dim];
    else
      v = qkv[(size_t)(s * WSEG + key - PTOK) * NQKV + HEADS * DH + h * DH + dim];
    sKT[dim * KEYS + key] = v;
  }
  // zero probability buffer (pads key 80..95 with 0)
  for (int c = tid; c < (4 * 16 * KEYPAD) / 8; c += 128)
    *(uint4*)(&sP[0][0] + c * 8) = make_uint4(0, 0, 0, 0);

  __syncthreads();

  // Q fragments (16x64 per wave) straight from global
  const __bf16* qrowp = qkv + (size_t)(s * WSEG + wave * 16 + l) * NQKV + h * DH;
  v16bf aq[2];
#pragma unroll
  for (int kf = 0; kf < 2; ++kf) {
    v8bf lo = *(const v8bf*)(qrowp + kf * 32 + half * 8);
    v8bf hi = *(const v8bf*)(qrowp + kf * 32 + 16 + half * 8);
    aq[kf] = __builtin_shufflevector(lo, hi, 0, 1, 2, 3, 4, 5, 6, 7, 8, 9, 10,
                                     11, 12, 13, 14, 15);
  }

  // scores (16 x 80) = Q @ K^T, 5 key-tiles x 2 k-steps
  const v8f vzero = {};
  v8f st[5];
#pragma unroll
  for (int nt = 0; nt < 5; ++nt) {
    v8f a = vzero;
#pragma unroll
    for (int kf = 0; kf < 2; ++kf) {
      v16bf bfr;
#pragma unroll
      for (int t = 0; t < 16; ++t)
        bfr[t] = sKT[(kf * 32 + (t & 7) + 8 * half + 16 * (t >> 3)) * KEYS + nt * 16 + l];
      a = __builtin_amdgcn_wmma_f32_16x16x32_bf16(false, aq[kf], false, bfr,
                                                  (short)0, a, false, false);
    }
    st[nt] = a;
  }

  // top-left causal mask (key j <= query i), scale, row softmax in
  // accumulator layout (reduce inside 16-lane halves of the wave32)
#pragma unroll
  for (int v = 0; v < 8; ++v) {
    const int qrow = wave * 16 + v + 8 * half;
    float mx = -3.0e38f;
#pragma unroll
    for (int nt = 0; nt < 5; ++nt) {
      int key = nt * 16 + l;
      float sc = (key <= qrow) ? st[nt][v] * 0.125f : -__builtin_inff();
      st[nt][v] = sc;
      mx = fmaxf(mx, sc);
    }
#pragma unroll
    for (int off = 8; off > 0; off >>= 1) mx = fmaxf(mx, __shfl_xor(mx, off, 32));
    float sum = 0.f;
#pragma unroll
    for (int nt = 0; nt < 5; ++nt) {
      float e = __expf(st[nt][v] - mx);
      st[nt][v] = e;
      sum += e;
    }
#pragma unroll
    for (int off = 8; off > 0; off >>= 1) sum += __shfl_xor(sum, off, 32);
    const float inv = 1.0f / sum;
#pragma unroll
    for (int nt = 0; nt < 5; ++nt)
      sP[wave][(v + 8 * half) * KEYPAD + nt * 16 + l] = f2bf(st[nt][v] * inv);
  }

  __syncthreads();

  // out (16x64) = P(16x96) @ V(96x64)
  v8f ot[4];
#pragma unroll
  for (int nt = 0; nt < 4; ++nt) ot[nt] = vzero;
#pragma unroll
  for (int kf = 0; kf < 3; ++kf) {
    const __bf16* pr = &sP[wave][l * KEYPAD + kf * 32];
    v8bf lo = *(const v8bf*)(pr + half * 8);
    v8bf hi = *(const v8bf*)(pr + 16 + half * 8);
    v16bf ap = __builtin_shufflevector(lo, hi, 0, 1, 2, 3, 4, 5, 6, 7, 8, 9,
                                       10, 11, 12, 13, 14, 15);
#pragma unroll
    for (int nt = 0; nt < 4; ++nt) {
      v16bf bv;
#pragma unroll
      for (int t = 0; t < 16; ++t)
        bv[t] = sV[(kf * 32 + (t & 7) + 8 * half + 16 * (t >> 3)) * DH + nt * 16 + l];
      ot[nt] = __builtin_amdgcn_wmma_f32_16x16x32_bf16(false, ap, false, bv,
                                                       (short)0, ot[nt], false,
                                                       false);
    }
  }

  // write per-head slice of the (NTOK x 2048) attention output
#pragma unroll
  for (int nt = 0; nt < 4; ++nt) {
    int col = h * DH + nt * 16 + l;
#pragma unroll
    for (int v = 0; v < 8; ++v) {
      int tok = s * WSEG + wave * 16 + v + 8 * half;
      aout[(size_t)tok * DIMX + col] = f2bf(ot[nt][v]);
    }
  }
}

// --------------------------- launcher --------------------------------------
extern "C" void kernel_launch(void* const* d_in, const int* in_sizes, int n_in,
                              void* d_out, int out_size, void* d_ws,
                              size_t ws_size, hipStream_t stream) {
  (void)in_sizes; (void)n_in; (void)out_size; (void)ws_size;
  const float* seq    = (const float*)d_in[0];
  const float* w_norm = (const float*)d_in[1];
  const float* w_qkv  = (const float*)d_in[2];
  const float* w_out  = (const float*)d_in[3];
  const float* pm     = (const float*)d_in[4];
  float* out = (float*)d_out;

  char* ws = (char*)d_ws;
  size_t off = 0;
  auto carve = [&](size_t elems) {
    __bf16* p = (__bf16*)(ws + off);
    off += ((elems * 2 + 255) & ~(size_t)255);
    return p;
  };
  __bf16* x_bf    = carve((size_t)NTOK * DIMX);
  __bf16* wqkv_bf = carve((size_t)DIMX * NQKV);
  __bf16* wout_bf = carve((size_t)DIMX * DIMX);
  __bf16* pm_bf   = carve((size_t)2 * HEADS * PTOK * DH);
  __bf16* qkv_bf  = carve((size_t)NTOK * NQKV);
  __bf16* ao_bf   = carve((size_t)NTOK * DIMX);

  rmsnorm_cast_kernel<<<NTOK, 256, 0, stream>>>(seq, w_norm, x_bf);

  int n1 = DIMX * NQKV;
  cast_kernel<<<(n1 + 255) / 256, 256, 0, stream>>>(w_qkv, wqkv_bf, n1);
  int n2 = DIMX * DIMX;
  cast_kernel<<<(n2 + 255) / 256, 256, 0, stream>>>(w_out, wout_bf, n2);
  int n3 = 2 * HEADS * PTOK * DH;
  cast_kernel<<<(n3 + 255) / 256, 256, 0, stream>>>(pm, pm_bf, n3);

  gemm_bf16<true><<<dim3(NQKV / 128, NTOK / 128), 256, 0, stream>>>(
      x_bf, wqkv_bf, qkv_bf, NTOK, NQKV, DIMX);

  attn_kernel<<<dim3(HEADS, NTOK / WSEG), 128, 0, stream>>>(qkv_bf, pm_bf,
                                                            ao_bf);

  gemm_bf16<false><<<dim3(DIMX / 128, NTOK / 128), 256, 0, stream>>>(
      ao_bf, wout_bf, out, NTOK, DIMX, DIMX);
}
